// TernaryLinear_84619445666060
// MI455X (gfx1250) — matrix operand order repaired and checked by reading, not compiled, not run on hardware
//
#include <hip/hip_runtime.h>

// ---------------------------------------------------------------------------
// TernaryLinear on gfx1250: out = x @ ternary(W)^T + b
//   x : [16384, 4096] f32, W : [4096, 4096] f32, b : [4096] f32
// Strategy: Wq in {-1,0,1} is exact in f16; split x = hi + lo (both f16) so
// every product is exact and v_wmma_f32_16x16x32_f16 accumulates in f32.
// ---------------------------------------------------------------------------

typedef __attribute__((ext_vector_type(16))) _Float16 v16h;
typedef __attribute__((ext_vector_type(8)))  float    v8f;

#define TL_M 16384
#define TL_K 4096
#define TL_N 4096

#define BM 128
#define BN 128
#define BK 64
#define LDS_STRIDE 72          // BK + 8 halves of padding (144B rows, 16B aligned)
#define RED_BLOCKS 1024

union FragU {
    uint4 q[2];
    v16h  v;
};

// ------------------------- alpha = mean(|W|) -------------------------------

__global__ __launch_bounds__(256)
void tl_reduce_partial(const float* __restrict__ W, float* __restrict__ part) {
    __shared__ float red[256];
    const long long total = (long long)TL_N * TL_K;
    float s = 0.0f;
    for (long long i = (long long)blockIdx.x * 256 + threadIdx.x; i < total;
         i += (long long)gridDim.x * 256)
        s += fabsf(W[i]);
    red[threadIdx.x] = s;
    __syncthreads();
    for (int off = 128; off > 0; off >>= 1) {
        if ((int)threadIdx.x < off) red[threadIdx.x] += red[threadIdx.x + off];
        __syncthreads();
    }
    if (threadIdx.x == 0) part[blockIdx.x] = red[0];
}

__global__ __launch_bounds__(256)
void tl_reduce_final(float* __restrict__ part) {
    __shared__ float red[256];
    float s = 0.0f;
    for (int i = threadIdx.x; i < RED_BLOCKS; i += 256) s += part[i];
    red[threadIdx.x] = s;
    __syncthreads();
    for (int off = 128; off > 0; off >>= 1) {
        if ((int)threadIdx.x < off) red[threadIdx.x] += red[threadIdx.x + off];
        __syncthreads();
    }
    if (threadIdx.x == 0) {
        float alpha = red[0] / (float)((long long)TL_N * TL_K);
        part[RED_BLOCKS] = 0.5f * alpha;      // quantization threshold
    }
}

// ------------------------------- GEMM --------------------------------------
// Block: 256 threads = 8 waves (wave32). Wave grid 4 (M) x 2 (N):
//   each wave owns a 32x64 patch = 2x4 tiles of 16x16, 8 v8f accumulators.
// Per K-step of 32: 2 (m-tiles) x 4 (n-tiles) x 2 (hi/lo) = 16 WMMAs.

__global__ __launch_bounds__(256)
void tl_gemm(const float* __restrict__ x, const float* __restrict__ W,
             const float* __restrict__ bias, float* __restrict__ out,
             const float* __restrict__ wsf) {
    __shared__ _Float16 sXhi[BM][LDS_STRIDE];
    __shared__ _Float16 sXlo[BM][LDS_STRIDE];
    __shared__ _Float16 sWq [BN][LDS_STRIDE];

    const float thr = wsf[RED_BLOCKS];
    const int tid     = threadIdx.x;
    const int lane    = tid & 31;
    const int wave    = tid >> 5;
    const int wm      = wave & 3;     // 0..3 : 32-row band of M
    const int wn      = wave >> 2;    // 0..1 : 64-col band of N
    const int halfSel = lane >> 4;    // lane group (0: lanes 0-15, 1: 16-31)
    const int l16     = lane & 15;
    const int blockM  = blockIdx.y * BM;
    const int blockN  = blockIdx.x * BN;

    v8f zero = {};
    v8f acc[2][4];
#pragma unroll
    for (int i = 0; i < 2; ++i)
#pragma unroll
        for (int j = 0; j < 4; ++j) acc[i][j] = zero;

    for (int kc = 0; kc < TL_K; kc += BK) {
        // ---- stage x (-> hi/lo f16) and quantized W into LDS -------------
        // 128 rows x 64 cols = 2048 float4 chunks each; 8 chunks per thread.
#pragma unroll
        for (int c = 0; c < 8; ++c) {
            const int chunk = tid + c * 256;          // 0..2047
            const int row   = chunk >> 4;             // 0..127
            const int col   = (chunk & 15) << 2;      // 0..60

            const float4 xv = *reinterpret_cast<const float4*>(
                &x[(long long)(blockM + row) * TL_K + kc + col]);
            const float xe[4] = {xv.x, xv.y, xv.z, xv.w};
#pragma unroll
            for (int e = 0; e < 4; ++e) {
                const float f  = xe[e];
                const _Float16 hi = (_Float16)f;              // RNE
                const _Float16 lo = (_Float16)(f - (float)hi); // exact residual
                sXhi[row][col + e] = hi;
                sXlo[row][col + e] = lo;
            }

            const float4 wv = *reinterpret_cast<const float4*>(
                &W[(long long)(blockN + row) * TL_K + kc + col]);
            const float we[4] = {wv.x, wv.y, wv.z, wv.w};
#pragma unroll
            for (int e = 0; e < 4; ++e) {
                const float w = we[e];
                const float q = (fabsf(w) > thr) ? (w > 0.0f ? 1.0f : -1.0f) : 0.0f;
                sWq[row][col + e] = (_Float16)q;              // exact in f16
            }
        }

        // prefetch next K chunk while this one is consumed
        if (kc + BK < TL_K) {
            __builtin_prefetch(&x[(long long)(blockM + (tid >> 1)) * TL_K +
                                  kc + BK + (tid & 1) * 32], 0, 0);
            __builtin_prefetch(&W[(long long)(blockN + (tid >> 1)) * TL_K +
                                  kc + BK + (tid & 1) * 32], 0, 0);
        }
        __syncthreads();

        // ---- WMMA over the staged 64-wide K chunk ------------------------
#pragma unroll
        for (int kt = 0; kt < BK; kt += 32) {
            // A fragments (16-bit 16x32 layout):
            //   lane holds M = l16; halves 0..7  -> K = kt + halfSel*8 + h
            //                       halves 8..15 -> K = kt + 16 + halfSel*8 + h
            FragU ahi[2], alo[2];
#pragma unroll
            for (int i = 0; i < 2; ++i) {
                const int row = wm * 32 + i * 16 + l16;
                const int c0  = kt + halfSel * 8;
                ahi[i].q[0] = *reinterpret_cast<const uint4*>(&sXhi[row][c0]);
                ahi[i].q[1] = *reinterpret_cast<const uint4*>(&sXhi[row][c0 + 16]);
                alo[i].q[0] = *reinterpret_cast<const uint4*>(&sXlo[row][c0]);
                alo[i].q[1] = *reinterpret_cast<const uint4*>(&sXlo[row][c0 + 16]);
            }
            // B fragments (16-bit 32x16 layout):
            //   lane holds N = l16; halves 0..7  -> K = kt + halfSel*8 + h
            //                       halves 8..15 -> K = kt + 16 + halfSel*8 + h
#pragma unroll
            for (int j = 0; j < 4; ++j) {
                FragU bf;
                const int wrow = wn * 64 + j * 16 + l16;
                const int c0   = kt + halfSel * 8;
                bf.q[0] = *reinterpret_cast<const uint4*>(&sWq[wrow][c0]);
                bf.q[1] = *reinterpret_cast<const uint4*>(&sWq[wrow][c0 + 16]);
#pragma unroll
                for (int i = 0; i < 2; ++i) {
                    acc[i][j] = __builtin_amdgcn_wmma_f32_16x16x32_f16(
                        false, ahi[i].v, false, bf.v, (short)0, acc[i][j],
                        false, false);
                    acc[i][j] = __builtin_amdgcn_wmma_f32_16x16x32_f16(
                        false, alo[i].v, false, bf.v, (short)0, acc[i][j],
                        false, false);
                }
            }
        }
        __syncthreads();
    }

    // ---- epilogue: bias add + store (C layout: VGPR r -> M = halfSel*8+r,
    //      lane -> N = l16) ---------------------------------------------------
#pragma unroll
    for (int j = 0; j < 4; ++j) {
        const int col = blockN + wn * 64 + j * 16 + l16;
        const float bv = bias[col];
#pragma unroll
        for (int i = 0; i < 2; ++i) {
            const int mbase = blockM + wm * 32 + i * 16 + halfSel * 8;
#pragma unroll
            for (int r = 0; r < 8; ++r) {
                out[(long long)(mbase + r) * TL_N + col] = acc[i][j][r] + bv;
            }
        }
    }
}

// ------------------------------ launcher -----------------------------------

extern "C" void kernel_launch(void* const* d_in, const int* in_sizes, int n_in,
                              void* d_out, int out_size, void* d_ws, size_t ws_size,
                              hipStream_t stream) {
    (void)in_sizes; (void)n_in; (void)out_size; (void)ws_size;
    const float* x  = (const float*)d_in[0];
    const float* W  = (const float*)d_in[1];
    const float* b  = (const float*)d_in[2];
    float*       out = (float*)d_out;
    float*       wsf = (float*)d_ws;   // [0..1023] partials, [1024] threshold

    tl_reduce_partial<<<RED_BLOCKS, 256, 0, stream>>>(W, wsf);
    tl_reduce_final  <<<1,          256, 0, stream>>>(wsf);

    dim3 grid(TL_N / BN, TL_M / BM);   // (32, 128)
    tl_gemm<<<grid, 256, 0, stream>>>(x, W, b, out, wsf);
}
